// MultiHeadSelfAttentionEins_8375186227717
// MI455X (gfx1250) — matrix operand order repaired and checked by reading, not compiled, not run on hardware
//
#include <hip/hip_runtime.h>

// Problem constants (from reference)
#define Bn   4
#define Tn   2048
#define DIMn 1024
#define Hn   16
#define DHn  64
#define NQKV (3 * Hn * DHn)   // 3072
#define ROWS (Bn * Tn)        // 8192

typedef __attribute__((ext_vector_type(16))) __bf16 v16bf;
typedef __attribute__((ext_vector_type(8)))  __bf16 v8bf;
typedef __attribute__((ext_vector_type(8)))  float  v8f;

union FragBF { v16bf v; v8bf h[2]; };

__device__ __forceinline__ unsigned short bfbits(float f) {
  union { float f; unsigned u; } a; a.f = f;
  unsigned u = a.u;
  u += 0x7FFFu + ((u >> 16) & 1u);   // round-to-nearest-even
  return (unsigned short)(u >> 16);
}

// --- CDNA5 async copy: global -> LDS, tracked by ASYNCcnt --------------------
__device__ __forceinline__ void ld_async_b128(unsigned short* lds_dst,
                                              const unsigned short* gsrc) {
  // low 32 bits of a generic pointer to __shared__ == LDS byte offset
  unsigned ldsoff = (unsigned)(uintptr_t)lds_dst;
  asm volatile("global_load_async_to_lds_b128 %0, %1, off"
               :: "v"(ldsoff), "v"(gsrc)
               : "memory");
}
__device__ __forceinline__ void wait_async0() {
  asm volatile("s_wait_asynccnt 0x0" ::: "memory");
}

// ---------------------------------------------------------------------------
// Prep kernels: fp32 -> bf16 conversion, weight transposition to (N, K) bf16
// ---------------------------------------------------------------------------
__global__ void cvt_x_kernel(const float* __restrict__ x,
                             unsigned short* __restrict__ xb, int n) {
  int i = blockIdx.x * blockDim.x + threadIdx.x;
  if (i < n) xb[i] = bfbits(x[i]);
}

__global__ void tr_wqkv_kernel(const float* __restrict__ w,
                               unsigned short* __restrict__ wt) {
  int c = blockIdx.x;                      // 0..3071 (output column)
  for (int k = threadIdx.x; k < DIMn; k += blockDim.x)
    wt[(size_t)c * DIMn + k] = bfbits(w[(size_t)k * NQKV + c]);
}

__global__ void tr_w0_kernel(const float* __restrict__ w,
                             unsigned short* __restrict__ wt) {
  int c = blockIdx.x;                      // 0..1023
  for (int k = threadIdx.x; k < DIMn; k += blockDim.x)
    wt[(size_t)c * DIMn + k] = bfbits(w[(size_t)k * DIMn + c]);
}

// ---------------------------------------------------------------------------
// Shared GEMM body: block tile M=256 (8 waves x 32 rows) x N=64.
// B panel (64 x 32 bf16) staged to LDS by async b128 copies, double-buffered,
// rows padded to 80B (bank-conflict-free, 16B aligned).
// acc[2][4] = this wave's 32x64 C tile (8 WMMAs per 32-K step).
// ---------------------------------------------------------------------------
#define BSTR 40                 // padded B row stride in bf16 elems (80 bytes)
#define BBUF (64 * BSTR)

__device__ __forceinline__ void gemm_tile_32x64_staged(
    const unsigned short* __restrict__ a0,     // A + (m0+ln)*DIMn
    const unsigned short* __restrict__ a1,     // A + (m0+16+ln)*DIMn
    const unsigned short* __restrict__ bt_n0,  // B^T + n0*DIMn  (N-major rows)
    unsigned short* __restrict__ sB,           // [2][BBUF] bf16
    int tid, int half, int ln, v8f acc[2][4]) {
  const int srow = tid >> 2;                   // 0..63 : B^T row (output col)
  const int schk = tid & 3;                    // 0..3  : 8-elem chunk in K-panel

  // prologue: stage K-panel 0 into buffer 0
  ld_async_b128(sB + srow * BSTR + schk * 8,
                bt_n0 + (size_t)srow * DIMn + schk * 8);
  wait_async0();
  __syncthreads();

  int p = 0;
  for (int k0 = 0; k0 < DIMn; k0 += 32) {
    if (k0 + 32 < DIMn) {                      // prefetch next panel
      ld_async_b128(sB + (1 - p) * BBUF + srow * BSTR + schk * 8,
                    bt_n0 + (size_t)srow * DIMn + (k0 + 32) + schk * 8);
    }
    FragBF a[2];
    a[0].h[0] = *(const v8bf*)(const void*)(a0 + k0 + half * 16);
    a[0].h[1] = *(const v8bf*)(const void*)(a0 + k0 + half * 16 + 8);
    a[1].h[0] = *(const v8bf*)(const void*)(a1 + k0 + half * 16);
    a[1].h[1] = *(const v8bf*)(const void*)(a1 + k0 + half * 16 + 8);
    const unsigned short* bbuf = sB + p * BBUF;
#pragma unroll
    for (int nt = 0; nt < 4; ++nt) {
      FragBF bf;
      bf.v = *(const v16bf*)(const void*)(bbuf + (nt * 16 + ln) * BSTR + half * 16);
#pragma unroll
      for (int mt = 0; mt < 2; ++mt)
        acc[mt][nt] = __builtin_amdgcn_wmma_f32_16x16x32_bf16(
            false, a[mt].v, false, bf.v, (short)0, acc[mt][nt], false, false);
    }
    wait_async0();                             // next panel landed in LDS
    __syncthreads();                           // all waves done reading buf p
    p ^= 1;
  }
}

// ---------------------------------------------------------------------------
// QKV projection: (8192 x 1024) x (1024 x 3072) with bf16 WMMA,
// scatter to Q (b,h,t,d) [pre-scaled by DH^-0.5], K (b,h,t,d), V^T (b,h,d,t)
// ---------------------------------------------------------------------------
__global__ void gemm_qkv_kernel(const unsigned short* __restrict__ xb,
                                const unsigned short* __restrict__ wt,
                                unsigned short* __restrict__ qb,
                                unsigned short* __restrict__ kb,
                                unsigned short* __restrict__ vt) {
  __shared__ unsigned short sB[2][BBUF];

  const int tid  = threadIdx.x;
  const int lane = tid & 31;
  const int half = lane >> 4;
  const int ln   = lane & 15;
  const int wave = tid >> 5;

  const int bm = blockIdx.x / (NQKV / 64);     // 32 M-blocks
  const int bn = blockIdx.x % (NQKV / 64);     // 48 N-blocks
  const int m0 = bm * 256 + wave * 32;
  const int n0 = bn * 64;

  v8f acc[2][4] = {};
  gemm_tile_32x64_staged(xb + (size_t)(m0 + ln) * DIMn,
                         xb + (size_t)(m0 + 16 + ln) * DIMn,
                         wt + (size_t)n0 * DIMn,
                         &sB[0][0], tid, half, ln, acc);

#pragma unroll
  for (int mt = 0; mt < 2; ++mt) {
#pragma unroll
    for (int nt = 0; nt < 4; ++nt) {
#pragma unroll
      for (int r = 0; r < 8; ++r) {
        int row = m0 + mt * 16 + r + 8 * half;
        int c   = n0 + nt * 16 + ln;
        int h   = c & 15;
        int kk  = (c >> 4) % 3;
        int d   = c / 48;
        int b   = row >> 11;                   // /2048
        int t   = row & 2047;
        size_t bh = (size_t)b * Hn + h;
        float v = acc[mt][nt][r];
        if (kk == 0)      qb[(bh * Tn + t) * DHn + d] = bfbits(v * 0.125f);
        else if (kk == 1) kb[(bh * Tn + t) * DHn + d] = bfbits(v);
        else              vt[(bh * DHn + d) * Tn + t] = bfbits(v);
      }
    }
  }
}

// ---------------------------------------------------------------------------
// Flash attention: causal + relative-position bias, bf16 WMMA, fp32 softmax.
// Block = one (b,h) x 128-row chunk; 8 waves x 16-row query tiles.
// K (32x64) and V^T (64x32) panels async-staged to LDS, double-buffered,
// shared by all waves; uniform j-loop to the block's causal max (each wave's
// causal range is a prefix), compute predicated per wave.
// ---------------------------------------------------------------------------
#define KSTR 72                 // padded K-panel row stride (144 bytes)
#define KBUF (32 * KSTR)
#define VSTR 40                 // padded V-panel row stride (80 bytes)
#define VBUF (64 * VSTR)
#define PSTR 40                 // padded P-scratch row stride

__global__ void attn_kernel(const unsigned short* __restrict__ qb,
                            const unsigned short* __restrict__ kb,
                            const unsigned short* __restrict__ vt,
                            const float* __restrict__ rel_bias,
                            unsigned short* __restrict__ aout) {
  __shared__ float sbias[Tn];              // rel_bias[j-i+T-1, h], idx 0..2047
  __shared__ unsigned short sK[2][KBUF];
  __shared__ unsigned short sV[2][VBUF];
  __shared__ unsigned short sP[8][16 * PSTR];

  const int tid  = threadIdx.x;
  const int lane = tid & 31;
  const int half = lane >> 4;
  const int ln   = lane & 15;
  const int wave = tid >> 5;

  const int bh    = blockIdx.x >> 4;       // 64 (b,h) pairs
  const int chunk = blockIdx.x & 15;       // 16 chunks of 128 rows
  const int h     = bh & (Hn - 1);
  const int b     = bh >> 4;
  const int i0    = chunk * 128 + wave * 16;

  const unsigned short* kbh = kb + (size_t)bh * Tn * DHn;
  const unsigned short* vbh = vt + (size_t)bh * DHn * Tn;

  // stage per-head bias slice (causal => only indices 0..T-1 are reachable)
  for (int p = tid; p < Tn; p += blockDim.x)
    sbias[p] = rel_bias[(size_t)p * Hn + h];
  __syncthreads();

  // Q fragments (A layout), K-dim = 64 split into two 32-chunks
  FragBF aq[2];
  {
    const unsigned short* qrow = qb + ((size_t)bh * Tn + i0 + ln) * DHn;
#pragma unroll
    for (int kc = 0; kc < 2; ++kc) {
      aq[kc].h[0] = *(const v8bf*)(const void*)(qrow + kc * 32 + half * 16);
      aq[kc].h[1] = *(const v8bf*)(const void*)(qrow + kc * 32 + half * 16 + 8);
    }
  }

  float m[8], l[8];
#pragma unroll
  for (int r = 0; r < 8; ++r) { m[r] = -__builtin_inff(); l[r] = 0.f; }
  v8f o[4] = {};

  // staging thread roles
  const int krow = tid >> 3, kchk = tid & 7;   // K panel: 32 rows x 4 chunks... (8 chunks of 8)
  const int vrow = tid >> 2, vchk = tid & 3;   // V panel: 64 rows x 4 chunks

  // prologue: stage jb = 0 into buffer 0
  ld_async_b128(&sK[0][krow * KSTR + kchk * 8],
                kbh + (size_t)krow * DHn + kchk * 8);
  ld_async_b128(&sV[0][vrow * VSTR + vchk * 8],
                vbh + (size_t)vrow * Tn + vchk * 8);
  wait_async0();
  __syncthreads();

  const int jmax = chunk * 128 + 128;          // block-wide causal bound
  int p = 0;
  for (int jb = 0; jb < jmax; jb += 32) {
    if (jb + 32 < jmax) {                      // prefetch next K/V panels
      ld_async_b128(&sK[1 - p][krow * KSTR + kchk * 8],
                    kbh + (size_t)(jb + 32 + krow) * DHn + kchk * 8);
      ld_async_b128(&sV[1 - p][vrow * VSTR + vchk * 8],
                    vbh + (size_t)vrow * Tn + (jb + 32) + vchk * 8);
    }

    if (jb < i0 + 16) {                        // this wave's causal range
      // S = Q K^T for a 16x32 score block (two 16x16 N-subtiles)
      v8f s[2] = {};
#pragma unroll
      for (int js = 0; js < 2; ++js) {
#pragma unroll
        for (int kc = 0; kc < 2; ++kc) {
          FragBF bk;
          bk.v = *(const v16bf*)(const void*)
              (&sK[p][(js * 16 + ln) * KSTR + kc * 32 + half * 16]);
          s[js] = __builtin_amdgcn_wmma_f32_16x16x32_bf16(
              false, aq[kc].v, false, bk.v, (short)0, s[js], false, false);
        }
      }

      // softmax update (rows striped: row = r + 8*half, col = lane&15)
      float pv0[8], pv1[8];
#pragma unroll
      for (int r = 0; r < 8; ++r) {
        int gi  = i0 + r + 8 * half;
        int gj0 = jb + ln;
        int gj1 = jb + 16 + ln;
        float v0 = (gj0 <= gi) ? s[0][r] + sbias[gj0 - gi + (Tn - 1)]
                               : -__builtin_inff();
        float v1 = (gj1 <= gi) ? s[1][r] + sbias[gj1 - gi + (Tn - 1)]
                               : -__builtin_inff();
        float mx = fmaxf(v0, v1);
#pragma unroll
        for (int d = 1; d < 16; d <<= 1) mx = fmaxf(mx, __shfl_xor(mx, d, 32));
        float mn    = fmaxf(m[r], mx);
        float alpha = __expf(m[r] - mn);
        float p0 = __expf(v0 - mn);
        float p1 = __expf(v1 - mn);
        float rs = p0 + p1;
#pragma unroll
        for (int d = 1; d < 16; d <<= 1) rs += __shfl_xor(rs, d, 32);
        l[r] = l[r] * alpha + rs;
        m[r] = mn;
        pv0[r] = p0; pv1[r] = p1;
#pragma unroll
        for (int nt = 0; nt < 4; ++nt) o[nt][r] *= alpha;
      }

      // transpose P (C layout) -> A layout via per-wave LDS scratch
      unsigned short* pw = sP[wave];
#pragma unroll
      for (int r = 0; r < 8; ++r) {
        int row = r + 8 * half;
        pw[row * PSTR + ln]      = bfbits(pv0[r]);
        pw[row * PSTR + 16 + ln] = bfbits(pv1[r]);
      }
      FragBF ap;
      ap.h[0] = *(const v8bf*)(const void*)&pw[ln * PSTR + half * 16];
      ap.h[1] = *(const v8bf*)(const void*)&pw[ln * PSTR + half * 16 + 8];

      // O += P V  (V panel in LDS, transposed: contiguous B-fragments)
#pragma unroll
      for (int nt = 0; nt < 4; ++nt) {
        FragBF bv;
        bv.v = *(const v16bf*)(const void*)
            (&sV[p][(nt * 16 + ln) * VSTR + half * 16]);
        o[nt] = __builtin_amdgcn_wmma_f32_16x16x32_bf16(
            false, ap.v, false, bv.v, (short)0, o[nt], false, false);
      }
    }

    wait_async0();                             // next panels landed
    __syncthreads();                           // all waves done with buf p
    p ^= 1;
  }

  // epilogue: normalize and write (b, t, h*64+d) bf16
#pragma unroll
  for (int nt = 0; nt < 4; ++nt) {
#pragma unroll
    for (int r = 0; r < 8; ++r) {
      int gi  = i0 + r + 8 * half;
      int col = h * DHn + nt * 16 + ln;
      float v = o[nt][r] / l[r];
      aout[((size_t)b * Tn + gi) * DIMn + col] = bfbits(v);
    }
  }
}

// ---------------------------------------------------------------------------
// Output projection: (8192 x 1024) bf16 x W0^T (1024 x 1024) bf16 -> fp32
// ---------------------------------------------------------------------------
__global__ void gemm_out_kernel(const unsigned short* __restrict__ ab,
                                const unsigned short* __restrict__ w0t,
                                float* __restrict__ out) {
  __shared__ unsigned short sB[2][BBUF];

  const int tid  = threadIdx.x;
  const int lane = tid & 31;
  const int half = lane >> 4;
  const int ln   = lane & 15;
  const int wave = tid >> 5;

  const int bm = blockIdx.x / (DIMn / 64);     // 32 M-blocks
  const int bn = blockIdx.x % (DIMn / 64);     // 16 N-blocks
  const int m0 = bm * 256 + wave * 32;
  const int n0 = bn * 64;

  v8f acc[2][4] = {};
  gemm_tile_32x64_staged(ab + (size_t)(m0 + ln) * DIMn,
                         ab + (size_t)(m0 + 16 + ln) * DIMn,
                         w0t + (size_t)n0 * DIMn,
                         &sB[0][0], tid, half, ln, acc);

#pragma unroll
  for (int mt = 0; mt < 2; ++mt) {
#pragma unroll
    for (int nt = 0; nt < 4; ++nt) {
#pragma unroll
      for (int r = 0; r < 8; ++r) {
        int row = m0 + mt * 16 + r + 8 * half;
        int col = n0 + nt * 16 + ln;
        out[(size_t)row * DIMn + col] = acc[mt][nt][r];
      }
    }
  }
}

// ---------------------------------------------------------------------------
// Host launcher
// ---------------------------------------------------------------------------
extern "C" void kernel_launch(void* const* d_in, const int* in_sizes, int n_in,
                              void* d_out, int out_size, void* d_ws, size_t ws_size,
                              hipStream_t stream) {
  const float* x     = (const float*)d_in[0];
  const float* Wqkv  = (const float*)d_in[1];
  const float* W0    = (const float*)d_in[2];
  const float* relb  = (const float*)d_in[3];
  float*       out   = (float*)d_out;
  char*        ws    = (char*)d_ws;

  // Workspace layout (bytes)
  const size_t SZ_XB    = (size_t)ROWS * DIMn * 2;        // 16 MB
  const size_t SZ_WQKVT = (size_t)NQKV * DIMn * 2;        //  6 MB
  const size_t SZ_W0T   = (size_t)DIMn * DIMn * 2;        //  2 MB
  const size_t SZ_QKV1  = (size_t)Bn * Hn * Tn * DHn * 2; // 16 MB each
  const size_t SZ_AOUT  = (size_t)ROWS * DIMn * 2;        // 16 MB

  size_t off = 0;
  unsigned short* xb    = (unsigned short*)(ws + off); off += SZ_XB;
  unsigned short* wqkvt = (unsigned short*)(ws + off); off += SZ_WQKVT;
  unsigned short* w0t   = (unsigned short*)(ws + off); off += SZ_W0T;
  unsigned short* qb    = (unsigned short*)(ws + off); off += SZ_QKV1;
  unsigned short* kb    = (unsigned short*)(ws + off); off += SZ_QKV1;
  unsigned short* vt    = (unsigned short*)(ws + off); off += SZ_QKV1;
  unsigned short* aout  = (unsigned short*)(ws + off); off += SZ_AOUT;
  (void)ws_size; (void)in_sizes; (void)n_in; (void)out_size;

  const int nx = ROWS * DIMn;
  cvt_x_kernel  <<<(nx + 255) / 256, 256, 0, stream>>>(x, xb, nx);
  tr_wqkv_kernel<<<NQKV, 256, 0, stream>>>(Wqkv, wqkvt);
  tr_w0_kernel  <<<DIMn, 256, 0, stream>>>(W0, w0t);

  // 32 M-blocks * 48 N-blocks, 8 waves/block (256x64 block tile)
  gemm_qkv_kernel<<<1536, 256, 0, stream>>>(xb, wqkvt, qb, kb, vt);

  // 64 (b,h) * 16 chunks = 1024 blocks, 8 query-tiles/block
  attn_kernel<<<1024, 256, 0, stream>>>(qb, kb, vt, relb, aout);

  // 32 M-blocks * 16 N-blocks, 8 waves/block
  gemm_out_kernel<<<512, 256, 0, stream>>>(aout, w0t, out);
}
